// KoLeoLoss_6622839570985
// MI455X (gfx1250) — matrix operand order repaired and checked by reading, not compiled, not run on hardware
//
#include <hip/hip_runtime.h>
#include <hip/hip_bf16.h>
#include <math.h>

// ---------------------------------------------------------------------------
// KoLeo loss on MI455X (gfx1250, wave32, WMMA + TDM).
//   N = 8192 rows, D = 1024 cols, fp32 input, scalar fp32 output.
// Search GEMM in f16 WMMA with TDM-staged LDS B tiles; fp32 refinement.
// ---------------------------------------------------------------------------

#define KO_N 8192
#define KO_D 1024
#define KO_EPS 1e-8f
#define KO_CHUNKS 16                       // column chunks for the argmax GEMM
#define KO_CHUNK_COLS (KO_N / KO_CHUNKS)   // 512
#define KO_ROWS_PER_BLK 128                // 8 waves * 16 rows

// B tile staged by TDM: 64 columns (rows of xh) x 128 K f16, padded rows.
#define KO_KSLAB 128
#define KO_NSLAB (KO_D / KO_KSLAB)         // 8
#define KO_BROW_BYTES 272                  // 256B data + 16B TDM pad -> bank-friendly
#define KO_BBUF_BYTES (64 * KO_BROW_BYTES) // 17408 per buffer

typedef __attribute__((ext_vector_type(16))) _Float16 v16h;
typedef __attribute__((ext_vector_type(8)))  _Float16 v8h;
typedef __attribute__((ext_vector_type(8)))  float    v8f;
typedef __attribute__((ext_vector_type(4)))  unsigned int v4u;
typedef __attribute__((ext_vector_type(8)))  int      v8i;
typedef __attribute__((ext_vector_type(4)))  int      v4i;

// -------------------------- kernel 1: normalize ----------------------------
__global__ void koleo_normalize(const float* __restrict__ in,
                                _Float16* __restrict__ xh,
                                float* __restrict__ inv,
                                float* __restrict__ s2) {
    const int row = blockIdx.x;
    const int tid = threadIdx.x;
    const float* rp = in + (size_t)row * KO_D;

    float vals[4];
    float acc = 0.0f;
#pragma unroll
    for (int s = 0; s < 4; ++s) {
        float v = rp[tid + 256 * s];
        vals[s] = v;
        acc += v * v;
    }
    __shared__ float red[256];
    __shared__ float sinv;
    red[tid] = acc;
    __syncthreads();
#pragma unroll
    for (int st = 128; st > 0; st >>= 1) {
        if (tid < st) red[tid] += red[tid + st];
        __syncthreads();
    }
    if (tid == 0) {
        float norm = sqrtf(red[0]);
        float iv = 1.0f / (norm + KO_EPS);
        inv[row] = iv;
        float sn = norm * iv;          // ||normalized row||
        s2[row] = sn * sn;
        sinv = iv;
    }
    __syncthreads();
    float iv = sinv;
#pragma unroll
    for (int s = 0; s < 4; ++s)
        xh[(size_t)row * KO_D + tid + 256 * s] = (_Float16)(vals[s] * iv);
}

// ------------------- fragment loaders (CDNA5 VGPR layouts) -----------------
// A (16x32 f16): lanes 0-15 hold K {0..7, 16..23}; lanes 16-31 hold {8..15, 24..31}
__device__ __forceinline__ v16h koleo_load_a(const _Float16* __restrict__ p, int h) {
    v8h lo = *reinterpret_cast<const v8h*>(p + 8 * h);
    v8h hi = *reinterpret_cast<const v8h*>(p + 16 + 8 * h);
    return __builtin_shufflevector(lo, hi, 0, 1, 2, 3, 4, 5, 6, 7,
                                           8, 9, 10, 11, 12, 13, 14, 15);
}
// B (32x16 f16 = X^T tile) from the TDM-staged LDS tile. Row n = output col,
// 272B pitch; K sub-chunk kk at byte kk*64; lanes 0-15: K {0..7, 16..23}, 16-31: {8..15, 24..31}.
__device__ __forceinline__ v16h koleo_load_b_lds(const unsigned char* __restrict__ base,
                                                 int n, int kk, int h) {
    const _Float16* p = reinterpret_cast<const _Float16*>(
        base + n * KO_BROW_BYTES + kk * 64 + 16 * h);
    v8h lo = *reinterpret_cast<const v8h*>(p);
    v8h hi = *reinterpret_cast<const v8h*>(p + 16);
    return __builtin_shufflevector(lo, hi, 0, 1, 2, 3, 4, 5, 6, 7,
                                           8, 9, 10, 11, 12, 13, 14, 15);
}

__device__ __forceinline__ void koleo_track(const v8f& c, int rowBase, int col,
                                            float* __restrict__ rmax,
                                            int* __restrict__ rarg) {
#pragma unroll
    for (int v = 0; v < 8; ++v) {
        int row = rowBase + v;
        float val = (row == col) ? -2.0f : c[v];  // mask diagonal (dots >= -1)
        if (val > rmax[v]) { rmax[v] = val; rarg[v] = col; }
    }
}

// ---------------- TDM: async tensor load of one B slab to LDS --------------
// Tile: 64 rows x 128 f16 (256B/row), pad 16B after every 256B stored
// (pad_interval=5 -> 64 DWORDs, pad_amount=3 -> 4 DWORDs) => 272B LDS pitch.
__device__ __forceinline__ void koleo_tdm_load_b(unsigned long long gaddr,
                                                 unsigned int lds_off) {
    v4u g0;
    g0[0] = 1u;                                   // count=1 (valid descriptor)
    g0[1] = lds_off;                              // lds_addr
    g0[2] = (unsigned int)gaddr;                  // global_addr[31:0]
    g0[3] = (unsigned int)((gaddr >> 32) & 0x01FFFFFFull) | (2u << 30); // type=2
    v8i g1;
    g1[0] = (int)((1u << 16)    // data_size = 2 bytes
                | (1u << 20)    // pad_enable
                | (5u << 22)    // pad_interval: 64 DWORDs (256B)
                | (3u << 25));  // pad_amount: 4 DWORDs (16B)
    g1[1] = (int)(((unsigned)KO_D & 0xFFFFu) << 16);                 // tensor_dim0 lo
    g1[2] = (int)((((unsigned)KO_D >> 16) & 0xFFFFu)
                | (((unsigned)KO_N & 0xFFFFu) << 16));               // dim0 hi | dim1 lo
    g1[3] = (int)((((unsigned)KO_N >> 16) & 0xFFFFu)
                | ((unsigned)KO_KSLAB << 16));                       // dim1 hi | tile_dim0
    g1[4] = 64;                                   // tile_dim1 = 64 rows, tile_dim2 = 0
    g1[5] = KO_D;                                 // tensor_dim0_stride lo32
    g1[6] = 0;                                    // stride hi | dim1_stride lo (unused 2D)
    g1[7] = 0;
    v4i z4 = {0, 0, 0, 0};
    v8i z8 = {0, 0, 0, 0, 0, 0, 0, 0};
    // amdgpu-toolchain (clang-23 / therock-10.0) 6-arg form:
    // (g0, g1, g2, g3, g4, cachepolicy)
    __builtin_amdgcn_tensor_load_to_lds(g0, g1, z4, z4, z8, 0);
}

// ----------------- kernel 2: WMMA GEMM + per-chunk argmax ------------------
// grid = (N/128, KO_CHUNKS), block = 256 (8 waves). Wave w owns 16 rows.
// B tiles (shared by all 8 waves) are TDM-staged into LDS, double-buffered.
__global__ void koleo_gemm_argmax(const _Float16* __restrict__ xh,
                                  float* __restrict__ pmax,
                                  int* __restrict__ pidx) {
    __shared__ __align__(16) unsigned char ldsB[2 * KO_BBUF_BYTES];

    const int lane = threadIdx.x & 31;
    const int wv   = threadIdx.x >> 5;
    const int r    = lane & 15;
    const int h    = lane >> 4;

    const int m0        = blockIdx.x * KO_ROWS_PER_BLK + wv * 16;
    const int chunkBase = blockIdx.y * KO_CHUNK_COLS;
    const int rowBase   = m0 + 8 * h;

    const unsigned int lds_base = (unsigned int)(size_t)(void*)ldsB;
    const unsigned long long xh_addr = (unsigned long long)(size_t)xh;

    float rmax[8];
    int   rarg[8];
#pragma unroll
    for (int v = 0; v < 8; ++v) { rmax[v] = -3.0e38f; rarg[v] = 0; }

    const _Float16* Ab = xh + (size_t)(m0 + r) * KO_D;

    for (int g = 0; g < KO_CHUNK_COLS / 64; ++g) {   // 8 groups of 64 cols
        const int n0 = chunkBase + g * 64;
        const unsigned long long gbase =
            xh_addr + ((unsigned long long)n0 * KO_D) * sizeof(_Float16);

        v8f c0 = {}, c1 = {}, c2 = {}, c3 = {};

        if (wv == 0)   // prologue: slab 0 -> buffer 0
            koleo_tdm_load_b(gbase, lds_base);

        for (int s = 0; s < KO_NSLAB; ++s) {
            if (wv == 0) {
                if (s + 1 < KO_NSLAB) {
                    koleo_tdm_load_b(gbase + (unsigned long long)(s + 1) * KO_KSLAB
                                               * sizeof(_Float16),
                                     lds_base + ((s + 1) & 1) * KO_BBUF_BYTES);
                    __builtin_amdgcn_s_wait_tensorcnt(1);  // slab s landed
                } else {
                    __builtin_amdgcn_s_wait_tensorcnt(0);
                }
            }
            __syncthreads();   // publish slab s to all waves

            const unsigned char* B = ldsB + (s & 1) * KO_BBUF_BYTES;
            const _Float16* Ak = Ab + s * KO_KSLAB;
            __builtin_prefetch(Ak + KO_KSLAB, 0, 1);
#pragma unroll
            for (int kk = 0; kk < KO_KSLAB / 32; ++kk) {
                v16h a  = koleo_load_a(Ak + kk * 32, h);
                v16h b0 = koleo_load_b_lds(B, r, kk, h);
                c0 = __builtin_amdgcn_wmma_f32_16x16x32_f16(false, a, false, b0,
                                                            (short)0, c0, false, false);
                v16h b1 = koleo_load_b_lds(B, 16 + r, kk, h);
                c1 = __builtin_amdgcn_wmma_f32_16x16x32_f16(false, a, false, b1,
                                                            (short)0, c1, false, false);
                v16h b2 = koleo_load_b_lds(B, 32 + r, kk, h);
                c2 = __builtin_amdgcn_wmma_f32_16x16x32_f16(false, a, false, b2,
                                                            (short)0, c2, false, false);
                v16h b3 = koleo_load_b_lds(B, 48 + r, kk, h);
                c3 = __builtin_amdgcn_wmma_f32_16x16x32_f16(false, a, false, b3,
                                                            (short)0, c3, false, false);
            }
            __syncthreads();   // all waves done with slab s before its buffer reuse
        }

        koleo_track(c0, rowBase, n0 + r,      rmax, rarg);
        koleo_track(c1, rowBase, n0 + 16 + r, rmax, rarg);
        koleo_track(c2, rowBase, n0 + 32 + r, rmax, rarg);
        koleo_track(c3, rowBase, n0 + 48 + r, rmax, rarg);
    }

    // Reduce across the 16 lanes of each half (off <= 8 keeps groups intact).
#pragma unroll
    for (int v = 0; v < 8; ++v) {
        float mv = rmax[v];
        int   mi = rarg[v];
#pragma unroll
        for (int off = 1; off < 16; off <<= 1) {
            float ov = __shfl_xor(mv, off, 32);
            int   oi = __shfl_xor(mi, off, 32);
            if (ov > mv || (ov == mv && oi < mi)) { mv = ov; mi = oi; }
        }
        if (r == 0) {
            int row = rowBase + v;
            pmax[(size_t)blockIdx.y * KO_N + row] = mv;
            pidx[(size_t)blockIdx.y * KO_N + row] = mi;
        }
    }
}

// -------- kernel 3: combine chunk partials, fp32 refine per row ------------
__global__ void koleo_refine(const float* __restrict__ in,
                             const float* __restrict__ inv,
                             const float* __restrict__ s2,
                             const float* __restrict__ pmax,
                             const int* __restrict__ pidx,
                             float* __restrict__ contrib) {
    const int i = blockIdx.x;
    const int tid = threadIdx.x;
    __shared__ float sdot[256];
    __shared__ int sj;

    if (tid == 0) {
        float best = -3.0e38f;
        int bj = 0x7fffffff;
        for (int c = 0; c < KO_CHUNKS; ++c) {
            float v = pmax[(size_t)c * KO_N + i];
            int j = pidx[(size_t)c * KO_N + i];
            if (v > best || (v == best && j < bj)) { best = v; bj = j; }
        }
        sj = bj;
    }
    __syncthreads();
    const int j = sj;

    float acc = 0.0f;
    const float* ri = in + (size_t)i * KO_D;
    const float* rj = in + (size_t)j * KO_D;
#pragma unroll
    for (int s = 0; s < 4; ++s) {
        int k = tid + 256 * s;
        acc += ri[k] * rj[k];
    }
    sdot[tid] = acc;
    __syncthreads();
#pragma unroll
    for (int st = 128; st > 0; st >>= 1) {
        if (tid < st) sdot[tid] += sdot[tid + st];
        __syncthreads();
    }
    if (tid == 0) {
        float dot = sdot[0] * inv[i] * inv[j];        // <x_i, x_j> exact fp32
        float d2 = s2[i] + s2[j] - 2.0f * dot;        // ||x_i - x_j||^2
        float d = sqrtf(fmaxf(d2, 0.0f));
        // reference: distances = norm + EPS; loss term = -log(distances + EPS)
        contrib[i] = -logf(d + 2.0f * KO_EPS);
    }
}

// --------------------- kernel 4: mean over rows ----------------------------
__global__ void koleo_final(const float* __restrict__ contrib,
                            float* __restrict__ out) {
    const int tid = threadIdx.x;
    __shared__ float s[256];
    float acc = 0.0f;
    for (int k = tid; k < KO_N; k += 256) acc += contrib[k];
    s[tid] = acc;
    __syncthreads();
#pragma unroll
    for (int st = 128; st > 0; st >>= 1) {
        if (tid < st) s[tid] += s[tid + st];
        __syncthreads();
    }
    if (tid == 0) out[0] = s[0] / (float)KO_N;
}

// ---------------------------------------------------------------------------
extern "C" void kernel_launch(void* const* d_in, const int* in_sizes, int n_in,
                              void* d_out, int out_size, void* d_ws, size_t ws_size,
                              hipStream_t stream) {
    (void)in_sizes; (void)n_in; (void)out_size; (void)ws_size;

    const float* in = (const float*)d_in[0];
    float* out = (float*)d_out;

    unsigned char* base = (unsigned char*)d_ws;
    size_t off = 0;
    auto take = [&](size_t bytes) -> void* {
        void* p = base + off;
        off = (off + bytes + 255) & ~(size_t)255;
        return p;
    };
    _Float16* xh   = (_Float16*)take((size_t)KO_N * KO_D * sizeof(_Float16)); // 16 MB
    float* inv     = (float*)take((size_t)KO_N * sizeof(float));
    float* s2      = (float*)take((size_t)KO_N * sizeof(float));
    float* pmax    = (float*)take((size_t)KO_CHUNKS * KO_N * sizeof(float));
    int*   pidx    = (int*)take((size_t)KO_CHUNKS * KO_N * sizeof(int));
    float* contrib = (float*)take((size_t)KO_N * sizeof(float));

    koleo_normalize<<<KO_N, 256, 0, stream>>>(in, xh, inv, s2);

    dim3 g2(KO_N / KO_ROWS_PER_BLK, KO_CHUNKS);   // 64 x 16 = 1024 blocks
    koleo_gemm_argmax<<<g2, 256, 0, stream>>>(xh, pmax, pidx);

    koleo_refine<<<KO_N, 256, 0, stream>>>(in, inv, s2, pmax, pidx, contrib);
    koleo_final<<<1, 256, 0, stream>>>(contrib, out);
}
